// Model_2078764171896
// MI455X (gfx1250) — matrix operand order repaired and checked by reading, not compiled, not run on hardware
//
#include <hip/hip_runtime.h>
#include <math.h>

// ---------------- model dims ----------------
#define BB     16
#define LL     96
#define NV     862
#define DM     512
#define DFF    1024
#define NH     8
#define HD     64
#define DFREQ  512
#define KFLEN  5000
#define KEYL   25      // -> y_key in_features = 50
#define KEYP   64      // padded 2*KEYL
#define FREQK  49      // rfft bins used by irfft
#define FREQP  64      // padded FREQK
#define NVP    864     // padded NV (K of P@V)
#define MROWS  (BB*NV) // 13792
#define TOTOUT ((long)BB*LL*NV)

typedef unsigned short bf16_t;
typedef __attribute__((ext_vector_type(16))) __bf16 v16bf;
typedef __attribute__((ext_vector_type(8)))  float  v8f;

__device__ inline float bf2f(bf16_t b){ unsigned int u=((unsigned int)b)<<16; float f; __builtin_memcpy(&f,&u,4); return f; }
__device__ inline bf16_t f2bf(float f){ unsigned int u; __builtin_memcpy(&u,&f,4); u = u + 0x7FFFu + ((u>>16)&1u); return (bf16_t)(u>>16); }

// ---------------- generic WMMA GEMM (B transposed: Bt is [N, ldb], ldb>=K) ----------------
// C[M,N] = act( A[M,K](bf16) * Bt[N,K]^T(bf16) + bias + resid ); C -> f32 and/or bf16.
// REQUIREMENTS (guaranteed by all call sites): K % 32 == 0, lda % 8 == 0, ldb % 8 == 0,
// base pointers 16B-aligned. Row-partial M/N tiles: staging row index is CLAMPED (the
// resulting garbage only feeds accumulator elements masked off in the epilogue);
// K-padding correctness is handled at the data level (zeroed pad rows/cols).
// Batched via blockIdx.z: z -> (zo=z/bdiv, zi=z%bdiv), ptr += zo*s?1 + zi*s?2.
#define TM 64
#define TN 128
#define TK 32
#define APAD 8

__global__ __launch_bounds__(256) void gemm_wmma(
    const bf16_t* __restrict__ A, long lda, long sA1, long sA2,
    const bf16_t* __restrict__ Bt, long ldb, long sB1, long sB2,
    float* __restrict__ Cf, bf16_t* __restrict__ Cb, long ldc, long sC1, long sC2,
    const float* __restrict__ bias,
    const float* __restrict__ resid, long ldr, long sR1, long sR2,
    int M, int N, int K, int act, int bdiv)
{
  __shared__ __align__(16) bf16_t As[TM][TK+APAD];
  __shared__ __align__(16) bf16_t Bs[TN][TK+APAD];

  long zo = (long)(blockIdx.z / (unsigned)bdiv);
  long zi = (long)(blockIdx.z % (unsigned)bdiv);
  A  += zo*sA1 + zi*sA2;
  Bt += zo*sB1 + zi*sB2;
  if (Cf)    Cf    += zo*sC1 + zi*sC2;
  if (Cb)    Cb    += zo*sC1 + zi*sC2;
  if (resid) resid += zo*sR1 + zi*sR2;

  const int m0  = blockIdx.y * TM;
  const int n0  = blockIdx.x * TN;
  const int tid = threadIdx.x;
  const int lane = tid & 31, wave = tid >> 5;
  const int wm = wave >> 2, wn = wave & 3;     // 2 x 4 wave grid -> 32x32 per wave

  // staging coordinates (uniform per thread across the whole K loop), rows clamped
  const int ar_ = tid >> 2, ac_ = (tid & 3) * 8;   // A: 64 rows x 32 cols, 8 bf16/thread
  const int bn_ = tid >> 1, bc_ = (tid & 1) * 16;  // B: 128 rows x 32 cols, 16 bf16/thread
  const long gmA = (long)((m0 + ar_) < M ? (m0 + ar_) : (M - 1));
  const long gnB = (long)((n0 + bn_) < N ? (n0 + bn_) : (N - 1));
  const bf16_t* aP = A  + gmA*lda + ac_;
  const bf16_t* bP = Bt + gnB*ldb + bc_;

  v8f acc[2][2] = {};

  // software pipeline: tile 0 into registers
  uint4 avR  = *(const uint4*)(aP);
  uint4 bv0R = *(const uint4*)(bP);
  uint4 bv1R = *(const uint4*)(bP + 8);

  for (int k0 = 0; k0 < K; k0 += TK){
    // ---- drain registers into LDS ----
    *(uint4*)&As[ar_][ac_]     = avR;
    *(uint4*)&Bs[bn_][bc_]     = bv0R;
    *(uint4*)&Bs[bn_][bc_ + 8] = bv1R;
    __syncthreads();

    // ---- issue next tile's global loads (overlap with WMMA below) ----
    int k1 = k0 + TK;
    if (k1 < K){                                   // scalar-uniform branch
      avR  = *(const uint4*)(aP + k1);
      bv0R = *(const uint4*)(bP + k1);
      bv1R = *(const uint4*)(bP + k1 + 8);
      if (k1 + TK < K){
        __builtin_prefetch(aP + k1 + TK, 0, 1);    // global_prefetch_b8 (warm L2)
        __builtin_prefetch(bP + k1 + TK, 0, 1);
      }
    }

    // ---- fragments (layouts per CDNA5 ISA 7.12.2) ----
    const int hl = lane >> 4, ml = lane & 15;
    union Frag { uint4 u[2]; v16bf v; };
    Frag af[2], bfg[2];
    #pragma unroll
    for (int i = 0; i < 2; ++i){
      int ar = wm*32 + i*16 + ml;
      af[i].u[0] = *(const uint4*)&As[ar][hl*8];
      af[i].u[1] = *(const uint4*)&As[ar][16 + hl*8];
    }
    #pragma unroll
    for (int j = 0; j < 2; ++j){
      int bn = wn*32 + j*16 + ml;
      bfg[j].u[0] = *(const uint4*)&Bs[bn][hl*8];
      bfg[j].u[1] = *(const uint4*)&Bs[bn][16 + hl*8];
    }
    #pragma unroll
    for (int i = 0; i < 2; ++i)
      #pragma unroll
      for (int j = 0; j < 2; ++j)
        acc[i][j] = __builtin_amdgcn_wmma_f32_16x16x32_bf16(
            false, af[i].v, false, bfg[j].v, (short)0, acc[i][j], false, false);
    __syncthreads();
  }

  // ---- epilogue ----
  const int hl = lane >> 4, nl = lane & 15;
  #pragma unroll
  for (int i = 0; i < 2; ++i)
    #pragma unroll
    for (int j = 0; j < 2; ++j){
      int mt = m0 + wm*32 + i*16;
      int nt = n0 + wn*32 + j*16;
      #pragma unroll
      for (int r = 0; r < 8; ++r){
        int m = mt + hl*8 + r;
        int n = nt + nl;
        if (m < M && n < N){
          float v = acc[i][j][r];
          if (bias)  v += bias[n];
          if (resid) v += resid[(long)m*ldr + n];
          if (act == 1)      v = (v >= 0.f) ? v : 0.01f*v;                       // leaky relu
          else if (act == 2) v = 0.5f*v*(1.f + erff(v*0.70710678118f));          // exact gelu
          else if (act == 3) v = (v > 0.f) ? v : 0.f;                            // relu
          if (Cf) Cf[(long)m*ldc + n] = v;
          if (Cb) Cb[(long)m*ldc + n] = f2bf(v);
        }
      }
    }
}

// ---------------- small kernels ----------------
// transpose-convert: src f32 [K,N] -> dst bf16 [N,KP], zero pad for k in [K,KP)
__global__ void cvtT_kernel(const float* __restrict__ src, bf16_t* __restrict__ dst,
                            int K, int N, int KP){
  long i = (long)blockIdx.x*blockDim.x + threadIdx.x;
  if (i >= (long)N*KP) return;
  int k = (int)(i % KP); long n = i / KP;
  dst[i] = (k < K) ? f2bf(src[(long)k*N + n]) : (bf16_t)0;
}

// transposed DFT bases: fwdT [KEYP, LL] (rows >= 50 zero), invT [LL, FREQP] (cols >= 49 zero)
__global__ void gen_dft_kernel(bf16_t* __restrict__ fwdT, bf16_t* __restrict__ invT){
  int t = blockIdx.x*blockDim.x + threadIdx.x;
  const float s = 0.10206207261596577f;      // 1/sqrt(96)
  const float w = 6.283185307179586f / 96.0f;
  if (t < KEYP*LL){
    int k = t / LL, l = t % LL;
    float v = 0.f;
    if (k < KEYL)            v =  cosf(w*(float)(k*l))*s;
    else if (k < 2*KEYL)     v = -sinf(w*(float)((k-KEYL)*l))*s;
    fwdT[t] = f2bf(v);
  } else if (t < KEYP*LL + LL*FREQP){
    int t2 = t - KEYP*LL;
    int l = t2 / FREQP, k = t2 % FREQP;
    float v = 0.f;
    if (k < FREQK){
      float c = (k == 0 || k == FREQK-1) ? 1.0f : 2.0f;
      v = s * c * cosf(w*(float)(k*l));
    }
    invT[t2] = f2bf(v);
  }
}

__global__ void revin_kernel(const float* __restrict__ x, const float* __restrict__ rw,
                             const float* __restrict__ rb, bf16_t* __restrict__ xn,
                             float* __restrict__ mean_o, float* __restrict__ std_o){
  int row = blockIdx.x*blockDim.x + threadIdx.x;
  if (row >= MROWS) return;
  int b = row / NV, n = row % NV;
  const float* base = x + (long)b*LL*NV + n;
  float s = 0.f;
  for (int l = 0; l < LL; ++l) s += base[(long)l*NV];
  float mu = s * (1.0f/LL);
  float v = 0.f;
  for (int l = 0; l < LL; ++l){ float d = base[(long)l*NV] - mu; v += d*d; }
  float sd = sqrtf(v*(1.0f/LL) + 1e-5f);
  mean_o[row] = mu; std_o[row] = sd;
  float w = rw[n], bsh = rb[n];
  for (int l = 0; l < LL; ++l)
    xn[(long)row*LL + l] = f2bf((base[(long)l*NV] - mu)/sd * w + bsh);
}

__global__ void trend_kernel(const bf16_t* __restrict__ xn, bf16_t* __restrict__ trend){
  int row = blockIdx.x*blockDim.x + threadIdx.x;
  if (row >= MROWS) return;
  float xr[LL];
  for (int l = 0; l < LL; ++l) xr[l] = bf2f(xn[(long)row*LL + l]);
  float s = 0.f;
  for (int d = -12; d <= 12; ++d){ int idx = d < 0 ? 0 : (d > LL-1 ? LL-1 : d); s += xr[idx]; }
  for (int l = 0; l < LL; ++l){
    trend[(long)row*LL + l] = f2bf(s * (1.0f/25.0f));
    int addi = l + 13; if (addi > LL-1) addi = LL-1;
    int remi = l - 12; if (remi < 0) remi = 0;
    s += xr[addi] - xr[remi];
  }
}

__global__ void gather_addsub_kernel(const float* __restrict__ src, const float* __restrict__ ft,
                                     const int* __restrict__ keys, bf16_t* __restrict__ out,
                                     int offset, float sign){
  long i = (long)blockIdx.x*blockDim.x + threadIdx.x;
  if (i >= (long)MROWS*DFREQ) return;
  long row = i >> 9;  int d = (int)(i & 511);
  int b = (int)(row / NV), n = (int)(row % NV);
  int gi = (keys[b] + offset + d) % KFLEN;
  out[i] = f2bf(src[i] + sign * ft[(long)gi*NV + n]);
}

// v [B*NV, DM] -> vT [B, DM, NVP] (pad cols j>=NV with zero)
__global__ void transpose_v_kernel(const bf16_t* __restrict__ v, bf16_t* __restrict__ vT){
  long i = (long)blockIdx.x*blockDim.x + threadIdx.x;
  if (i >= (long)BB*DM*NVP) return;
  int j = (int)(i % NVP); long t = i / NVP; int c = (int)(t % DM); int b = (int)(t / DM);
  vT[i] = (j < NV) ? v[((long)b*NV + j)*DM + c] : (bf16_t)0;
}

__device__ inline float block_reduce(float v, float* sm, int op){
  int t = threadIdx.x;
  sm[t] = v; __syncthreads();
  for (int s = 128; s > 0; s >>= 1){
    if (t < s){
      float a = sm[t], b = sm[t+s];
      sm[t] = (op == 0) ? a+b : (op == 1 ? fmaxf(a,b) : fminf(a,b));
    }
    __syncthreads();
  }
  float r = sm[0]; __syncthreads();
  return r;
}

__global__ void gating_kernel(const float* __restrict__ xc, const float* __restrict__ xng,
                              const float* __restrict__ noise, unsigned char* __restrict__ mask){
  __shared__ float sm[256];
  int row = blockIdx.x;                  // B*NV
  int i = row % NV;
  const float* xcr = xc  + (long)row*NV;
  const float* xnr = xng + (long)row*NV;
  float xv[4];
  float lmin = 3.4e38f, lmax = -3.4e38f;
  #pragma unroll
  for (int s = 0; s < 4; ++s){
    int j = threadIdx.x + s*256;
    float v = 0.f;
    if (j < NV){
      float w = xnr[j] + 0.01f;
      float nsp = (w > 0.f) ? -log1pf(expf(-w)) : (w - log1pf(expf(w)));  // -softplus(-w)
      v = xcr[j] + noise[j]*nsp;
      lmin = fminf(lmin, v); lmax = fmaxf(lmax, v);
    }
    xv[s] = v;
  }
  float mn = block_reduce(lmin, sm, 2);
  float mx = block_reduce(lmax, sm, 1);
  float inv = 1.0f / (mx - mn + 1e-6f);
  unsigned char* mrow = mask + (long)row*NV;
  #pragma unroll
  for (int s = 0; s < 4; ++s){
    int j = threadIdx.x + s*256;
    if (j < NV){
      float logit = (xv[s]-mn)*inv + ((j == i) ? 1.0f : 0.0f);
      mrow[j] = (logit < 0.5f) ? 1 : 0;
    }
  }
}

__global__ void layernorm_kernel(const float* __restrict__ x, const float* __restrict__ g,
                                 const float* __restrict__ b, float* __restrict__ yf,
                                 bf16_t* __restrict__ yb){
  __shared__ float sm[256];
  long row = blockIdx.x;
  const float* xr = x + row*DM;
  int c0 = threadIdx.x, c1 = threadIdx.x + 256;
  float v0 = xr[c0], v1 = xr[c1];
  float m = block_reduce(v0+v1, sm, 0) * (1.0f/DM);
  float d0 = v0-m, d1 = v1-m;
  float var = block_reduce(d0*d0 + d1*d1, sm, 0) * (1.0f/DM);
  float r = rsqrtf(var + 1e-5f);
  float o0 = d0*r*g[c0] + b[c0];
  float o1 = d1*r*g[c1] + b[c1];
  if (yf){ yf[row*DM+c0] = o0; yf[row*DM+c1] = o1; }
  if (yb){ yb[row*DM+c0] = f2bf(o0); yb[row*DM+c1] = f2bf(o1); }
}

// S rows (length NV) -> P rows (length NVP, cols >= NV zeroed)
__global__ void softmax_kernel(const float* __restrict__ S, const unsigned char* __restrict__ mask,
                               bf16_t* __restrict__ P){
  __shared__ float sm[256];
  long row = blockIdx.x;                 // bh*NV + i
  long bh = row / NV; int i = (int)(row % NV); int b = (int)(bh >> 3);
  const float* sr = S + row*(long)NV;
  const unsigned char* mr = mask + ((long)b*NV + i)*(long)NV;
  const float scale = 0.125f;            // 1/sqrt(64)
  float xv[4]; unsigned char mv[4];
  float lmax = -3.4e38f;
  #pragma unroll
  for (int s = 0; s < 4; ++s){
    int j = threadIdx.x + s*256;
    float v = -3.4e38f; unsigned char mm = 1;
    if (j < NV){ mm = mr[j]; v = mm ? -3.4e38f : sr[j]*scale; }
    xv[s] = v; mv[s] = mm;
    lmax = fmaxf(lmax, v);
  }
  float mx = block_reduce(lmax, sm, 1);
  float ev[4]; float lsum = 0.f;
  #pragma unroll
  for (int s = 0; s < 4; ++s){
    int j = threadIdx.x + s*256;
    float e = 0.f;
    if (j < NV && !mv[s]) e = expf(xv[s] - mx);
    ev[s] = e; lsum += e;
  }
  float tot = block_reduce(lsum, sm, 0);
  float rinv = 1.0f / tot;
  #pragma unroll
  for (int s = 0; s < 4; ++s){
    int j = threadIdx.x + s*256;
    if (j < NVP) P[row*(long)NVP + j] = (j < NV) ? f2bf(ev[s]*rinv) : (bf16_t)0;
  }
}

__global__ void finalize_kernel(const float* __restrict__ dec, const float* __restrict__ tloss,
                                const float* __restrict__ floss, const float* __restrict__ rw,
                                const float* __restrict__ rb, const float* __restrict__ mean,
                                const float* __restrict__ stdv, float* __restrict__ out){
  long idx = (long)blockIdx.x*blockDim.x + threadIdx.x;
  if (idx >= TOTOUT) return;
  int n = (int)(idx % NV); long t = idx / NV; int p = (int)(t % LL); int b = (int)(t / LL);
  long row = (long)b*NV + n;
  float sd = stdv[row], mu = mean[row];
  float denom = rw[n] + 1e-10f;          // revin_w + REVIN_EPS^2
  float bsh = rb[n];
  long src = row*LL + p;
  out[idx]            = (dec[src]   - bsh)/denom * sd + mu;
  out[TOTOUT + idx]   = (tloss[src] - bsh)/denom * sd + mu;
  out[2*TOTOUT + idx] = (floss[src] - bsh)/denom * sd + mu;
}

// ---------------- host side ----------------
static inline int cdiv_i(long a, long b){ return (int)((a + b - 1)/b); }

struct WSAlloc {
  char* p;
  template<typename T> T* take(long n){
    char* r = p;
    long bytes = n*(long)sizeof(T);
    p += (bytes + 255) & ~255L;
    return (T*)r;
  }
};

static void launch_gemm(hipStream_t st,
    const bf16_t* A, long lda, long sA1, long sA2,
    const bf16_t* Bt, long ldb, long sB1, long sB2,
    float* Cf, bf16_t* Cb, long ldc, long sC1, long sC2,
    const float* bias, const float* resid, long ldr, long sR1, long sR2,
    int M, int N, int K, int act, int batches, int bdiv)
{
  dim3 grid((unsigned)cdiv_i(N, TN), (unsigned)cdiv_i(M, TM), (unsigned)batches);
  gemm_wmma<<<grid,256,0,st>>>(A,lda,sA1,sA2,Bt,ldb,sB1,sB2,Cf,Cb,ldc,sC1,sC2,bias,resid,ldr,sR1,sR2,M,N,K,act,bdiv);
}

extern "C" void kernel_launch(void* const* d_in, const int* in_sizes, int n_in,
                              void* d_out, int out_size, void* d_ws, size_t ws_size,
                              hipStream_t stream) {
  (void)in_sizes; (void)n_in; (void)out_size; (void)ws_size;
  // -------- inputs (setup_inputs order) --------
  const float* x_enc   = (const float*)d_in[0];
  const float* gnoise  = (const float*)d_in[1];
  const float* revin_w = (const float*)d_in[2];
  const float* revin_b = (const float*)d_in[3];
  const float* fc2_w = (const float*)d_in[4];  const float* fc2_b = (const float*)d_in[5];
  const float* fc3_w = (const float*)d_in[6];  const float* fc3_b = (const float*)d_in[7];
  const float* fc4_w = (const float*)d_in[8];  const float* fc4_b = (const float*)d_in[9];
  const float* fc5_w = (const float*)d_in[10]; const float* fc5_b = (const float*)d_in[11];
  const float* tc_w1 = (const float*)d_in[12]; const float* tc_w2 = (const float*)d_in[13];
  const float* tn_w1 = (const float*)d_in[14]; const float* tn_w2 = (const float*)d_in[15];
  const float* lt_w1 = (const float*)d_in[16]; const float* lt_b1 = (const float*)d_in[17];
  const float* lt_w2 = (const float*)d_in[18]; const float* lt_b2 = (const float*)d_in[19];
  const float* yf_w  = (const float*)d_in[20]; const float* yf_b  = (const float*)d_in[21];
  const float* ls_w1 = (const float*)d_in[22]; const float* ls_b1 = (const float*)d_in[23];
  const float* ls_w2 = (const float*)d_in[24]; const float* ls_b2 = (const float*)d_in[25];
  const float* fco_w = (const float*)d_in[26]; const float* fco_b = (const float*)d_in[27];
  const float* ftab  = (const float*)d_in[28];
  const float* enc_wq = (const float*)d_in[29]; const float* enc_bq = (const float*)d_in[30];
  const float* enc_wk = (const float*)d_in[31]; const float* enc_bk = (const float*)d_in[32];
  const float* enc_wv = (const float*)d_in[33]; const float* enc_bv = (const float*)d_in[34];
  const float* enc_wo = (const float*)d_in[35]; const float* enc_bo = (const float*)d_in[36];
  const float* enc_c1w = (const float*)d_in[37]; const float* enc_c1b = (const float*)d_in[38];
  const float* enc_c2w = (const float*)d_in[39]; const float* enc_c2b = (const float*)d_in[40];
  const float* enc_g1 = (const float*)d_in[41]; const float* enc_be1 = (const float*)d_in[42];
  const float* enc_g2 = (const float*)d_in[43]; const float* enc_be2 = (const float*)d_in[44];
  const float* norm_g = (const float*)d_in[45]; const float* norm_b = (const float*)d_in[46];
  const int*   keys   = (const int*)d_in[47];
  float* out = (float*)d_out;

  const long M = MROWS;
  WSAlloc ws{ (char*)d_ws };

  // -------- workspace layout --------
  bf16_t* xn_bf   = ws.take<bf16_t>(M*LL);
  bf16_t* tr_bf   = ws.take<bf16_t>(M*LL);
  float*  meanv   = ws.take<float>(M);
  float*  stdv    = ws.take<float>(M);
  bf16_t* dftT    = ws.take<bf16_t>((long)KEYP*LL);   // [KEYP, LL]
  bf16_t* dinvT   = ws.take<bf16_t>((long)LL*FREQP);  // [LL, FREQP]
  // transposed bf16 weights: [N, KP]
  bf16_t* wfc2T = ws.take<bf16_t>((long)DM*LL);    bf16_t* wfc3T = ws.take<bf16_t>((long)LL*DM);
  bf16_t* wfc4T = ws.take<bf16_t>((long)DM*LL);    bf16_t* wfc5T = ws.take<bf16_t>((long)LL*DM);
  bf16_t* wtc1T = ws.take<bf16_t>((long)(DM/2)*DM);  bf16_t* wtc2T = ws.take<bf16_t>((long)NV*(DM/2));
  bf16_t* wtn1T = ws.take<bf16_t>((long)(DM/2)*DM);  bf16_t* wtn2T = ws.take<bf16_t>((long)NV*(DM/2));
  bf16_t* wlt1T = ws.take<bf16_t>((long)DM*LL);    bf16_t* wlt2T = ws.take<bf16_t>((long)DM*DM);
  bf16_t* wyfT  = ws.take<bf16_t>((long)DFREQ*KEYP);
  bf16_t* wls1T = ws.take<bf16_t>((long)DFREQ*DFREQ); bf16_t* wls2T = ws.take<bf16_t>((long)DFREQ*DFREQ);
  bf16_t* wfcoT = ws.take<bf16_t>((long)LL*(2*DM));
  bf16_t* wqT = ws.take<bf16_t>(2L*DM*DM);  bf16_t* wkT = ws.take<bf16_t>(2L*DM*DM);
  bf16_t* wvT = ws.take<bf16_t>(2L*DM*DM);  bf16_t* woT = ws.take<bf16_t>(2L*DM*DM);
  bf16_t* wc1T = ws.take<bf16_t>(2L*DFF*DM); bf16_t* wc2T = ws.take<bf16_t>(2L*DM*DFF);
  // activations
  bf16_t* h1bf  = ws.take<bf16_t>(M*DM);
  float*  t_f32 = ws.take<float>(M*DM);
  bf16_t* ykey0 = ws.take<bf16_t>(M*KEYP);
  float*  yk_f  = ws.take<float>(M*DFREQ);
  bf16_t* y2bf  = ws.take<bf16_t>(M*DFREQ);
  bf16_t* g1bf  = ws.take<bf16_t>(M*DFREQ);
  float*  yls_f = ws.take<float>(M*DFREQ);
  bf16_t* ybf   = ws.take<bf16_t>(M*DFREQ);
  bf16_t* xfret = ws.take<bf16_t>(M*LL);
  float*  xfre_f = ws.take<float>(M*DM);
  bf16_t* xfre_b = ws.take<bf16_t>(M*DM);
  float*  floss  = ws.take<float>(M*LL);
  float*  xb_f   = ws.take<float>(M*DM);
  bf16_t* xb_b   = ws.take<bf16_t>(M*DM);
  bf16_t* c1bf   = ws.take<bf16_t>(M*(DM/2));
  float*  xc_f   = ws.take<float>(M*NV);
  bf16_t* n1bf   = ws.take<bf16_t>(M*(DM/2));
  float*  xng_f  = ws.take<float>(M*NV);
  unsigned char* maskb = ws.take<unsigned char>((long)BB*NV*NV);
  bf16_t* qbf = ws.take<bf16_t>(M*DM);
  bf16_t* kbf = ws.take<bf16_t>(M*DM);
  bf16_t* vbf = ws.take<bf16_t>(M*DM);
  bf16_t* vTb = ws.take<bf16_t>((long)BB*DM*NVP);
  float*  Sbuf = ws.take<float>((long)BB*NH*NV*NV);
  bf16_t* Pbuf = ws.take<bf16_t>((long)BB*NH*NV*NVP);
  bf16_t* obf  = ws.take<bf16_t>(M*DM);
  float*  tmp  = ws.take<float>(M*DM);
  bf16_t* ffbf = ws.take<bf16_t>(M*DFF);
  float*  tloss = ws.take<float>(M*LL);
  float*  dec   = ws.take<float>(M*LL);

  // -------- weight f32 -> bf16 (transposed, K zero-padded) --------
  struct JobT { const float* s; bf16_t* d; int K, N, KP; };
  const JobT jobs[] = {
    {fc2_w, wfc2T, LL, DM, LL},   {fc3_w, wfc3T, DM, LL, DM},
    {fc4_w, wfc4T, LL, DM, LL},   {fc5_w, wfc5T, DM, LL, DM},
    {tc_w1, wtc1T, DM, DM/2, DM}, {tc_w2, wtc2T, DM/2, NV, DM/2},
    {tn_w1, wtn1T, DM, DM/2, DM}, {tn_w2, wtn2T, DM/2, NV, DM/2},
    {lt_w1, wlt1T, LL, DM, LL},   {lt_w2, wlt2T, DM, DM, DM},
    {yf_w,  wyfT,  2*KEYL, DFREQ, KEYP},
    {ls_w1, wls1T, DFREQ, DFREQ, DFREQ}, {ls_w2, wls2T, DFREQ, DFREQ, DFREQ},
    {fco_w, wfcoT, 2*DM, LL, 2*DM},
    {enc_wq,             wqT,             DM, DM, DM}, {enc_wq + DM*DM,   wqT + DM*DM,   DM, DM, DM},
    {enc_wk,             wkT,             DM, DM, DM}, {enc_wk + DM*DM,   wkT + DM*DM,   DM, DM, DM},
    {enc_wv,             wvT,             DM, DM, DM}, {enc_wv + DM*DM,   wvT + DM*DM,   DM, DM, DM},
    {enc_wo,             woT,             DM, DM, DM}, {enc_wo + DM*DM,   woT + DM*DM,   DM, DM, DM},
    {enc_c1w,            wc1T,            DM, DFF, DM}, {enc_c1w + DM*DFF, wc1T + (long)DFF*DM, DM, DFF, DM},
    {enc_c2w,            wc2T,            DFF, DM, DFF}, {enc_c2w + DFF*DM, wc2T + (long)DM*DFF, DFF, DM, DFF},
  };
  for (const JobT& j : jobs){
    long n = (long)j.N * j.KP;
    cvtT_kernel<<<cdiv_i(n,256),256,0,stream>>>(j.s, j.d, j.K, j.N, j.KP);
  }

  gen_dft_kernel<<<cdiv_i((long)KEYP*LL + (long)LL*FREQP,256),256,0,stream>>>(dftT, dinvT);

  // -------- RevIN + decomposition --------
  revin_kernel<<<cdiv_i(M,256),256,0,stream>>>(x_enc, revin_w, revin_b, xn_bf, meanv, stdv);
  trend_kernel<<<cdiv_i(M,256),256,0,stream>>>(xn_bf, tr_bf);

  // -------- frequency branch --------
  launch_gemm(stream, tr_bf,LL,0,0,  wlt1T,LL,0,0,  nullptr,h1bf,DM,0,0, lt_b1, nullptr,0,0,0, M,DM,LL, 1, 1,1);
  launch_gemm(stream, h1bf,DM,0,0,   wlt2T,DM,0,0,  t_f32,nullptr,DM,0,0, lt_b2, nullptr,0,0,0, M,DM,DM, 0, 1,1);
  launch_gemm(stream, xn_bf,LL,0,0,  dftT,LL,0,0,   nullptr,ykey0,KEYP,0,0, nullptr, nullptr,0,0,0, M,KEYP,LL, 0, 1,1);
  launch_gemm(stream, ykey0,KEYP,0,0, wyfT,KEYP,0,0, yk_f,nullptr,DFREQ,0,0, yf_b, nullptr,0,0,0, M,DFREQ,KEYP, 0, 1,1);
  gather_addsub_kernel<<<cdiv_i(M*DFREQ,256),256,0,stream>>>(yk_f, ftab, keys, y2bf, 0, -1.0f);
  launch_gemm(stream, y2bf,DFREQ,0,0, wls1T,DFREQ,0,0, nullptr,g1bf,DFREQ,0,0, ls_b1, nullptr,0,0,0, M,DFREQ,DFREQ, 1, 1,1);
  launch_gemm(stream, g1bf,DFREQ,0,0, wls2T,DFREQ,0,0, yls_f,nullptr,DFREQ,0,0, ls_b2, nullptr,0,0,0, M,DFREQ,DFREQ, 0, 1,1);
  gather_addsub_kernel<<<cdiv_i(M*DFREQ,256),256,0,stream>>>(yls_f, ftab, keys, ybf, DFREQ, 1.0f);
  launch_gemm(stream, ybf,DFREQ,0,0,  dinvT,FREQP,0,0, nullptr,xfret,LL,0,0, nullptr, nullptr,0,0,0, M,LL,FREQP, 0, 1,1);
  launch_gemm(stream, xfret,LL,0,0,   wfc4T,LL,0,0,  xfre_f,xfre_b,DM,0,0, fc4_b, t_f32,DM,0,0, M,DM,LL, 0, 1,1);
  launch_gemm(stream, xfre_b,DM,0,0,  wfc5T,DM,0,0,  floss,nullptr,LL,0,0, fc5_b, nullptr,0,0,0, M,LL,DM, 0, 1,1);

  // -------- time branch: embedding + gating mask --------
  launch_gemm(stream, xn_bf,LL,0,0,  wfc2T,LL,0,0,   xb_f,xb_b,DM,0,0, fc2_b, nullptr,0,0,0, M,DM,LL, 0, 1,1);
  launch_gemm(stream, xb_b,DM,0,0,   wtc1T,DM,0,0,   nullptr,c1bf,DM/2,0,0, nullptr, nullptr,0,0,0, M,DM/2,DM, 3, 1,1);
  launch_gemm(stream, c1bf,DM/2,0,0, wtc2T,DM/2,0,0, xc_f,nullptr,NV,0,0, nullptr, nullptr,0,0,0, M,NV,DM/2, 0, 1,1);
  launch_gemm(stream, xb_b,DM,0,0,   wtn1T,DM,0,0,   nullptr,n1bf,DM/2,0,0, nullptr, nullptr,0,0,0, M,DM/2,DM, 3, 1,1);
  launch_gemm(stream, n1bf,DM/2,0,0, wtn2T,DM/2,0,0, xng_f,nullptr,NV,0,0, nullptr, nullptr,0,0,0, M,NV,DM/2, 0, 1,1);
  gating_kernel<<<(unsigned)M,256,0,stream>>>(xc_f, xng_f, gnoise, maskb);

  // -------- encoder layers --------
  const long sBH = (long)NV*DM;     // per-batch stride in [M,512] activations
  for (int l = 0; l < 2; ++l){
    const bf16_t* wql = wqT + (long)l*DM*DM;  const bf16_t* wkl = wkT + (long)l*DM*DM;
    const bf16_t* wvl = wvT + (long)l*DM*DM;  const bf16_t* wol = woT + (long)l*DM*DM;
    launch_gemm(stream, xb_b,DM,0,0, wql,DM,0,0, nullptr,qbf,DM,0,0, enc_bq+l*DM, nullptr,0,0,0, M,DM,DM, 0, 1,1);
    launch_gemm(stream, xb_b,DM,0,0, wkl,DM,0,0, nullptr,kbf,DM,0,0, enc_bk+l*DM, nullptr,0,0,0, M,DM,DM, 0, 1,1);
    launch_gemm(stream, xb_b,DM,0,0, wvl,DM,0,0, nullptr,vbf,DM,0,0, enc_bv+l*DM, nullptr,0,0,0, M,DM,DM, 0, 1,1);
    transpose_v_kernel<<<cdiv_i((long)BB*DM*NVP,256),256,0,stream>>>(vbf, vTb);
    // S[b,h] = Q[b,:,h,:] * K[b,:,h,:]^T   (batched over z = b*8+h)
    launch_gemm(stream, qbf,DM, sBH,(long)HD, kbf,DM, sBH,(long)HD,
                Sbuf,nullptr,NV, (long)NH*NV*NV,(long)NV*NV,
                nullptr, nullptr,0,0,0, NV,NV,HD, 0, BB*NH, NH);
    softmax_kernel<<<(unsigned)((long)BB*NH*NV),256,0,stream>>>(Sbuf, maskb, Pbuf);
    // O[b,:,h,:] = P[b,h] * V[b,:,h,:]  via vT [B, DM, NVP]
    launch_gemm(stream, Pbuf,NVP, (long)NH*NV*NVP,(long)NV*NVP,
                vTb,NVP, (long)DM*NVP,(long)HD*NVP,
                nullptr,obf,DM, sBH,(long)HD,
                nullptr, nullptr,0,0,0, NV,HD,NVP, 0, BB*NH, NH);
    launch_gemm(stream, obf,DM,0,0, wol,DM,0,0, tmp,nullptr,DM,0,0, enc_bo+l*DM, xb_f,DM,0,0, M,DM,DM, 0, 1,1);
    layernorm_kernel<<<(unsigned)M,256,0,stream>>>(tmp, enc_g1+l*DM, enc_be1+l*DM, xb_f, xb_b);
    launch_gemm(stream, xb_b,DM,0,0, wc1T+(long)l*DFF*DM,DM,0,0, nullptr,ffbf,DFF,0,0, enc_c1b+l*DFF, nullptr,0,0,0, M,DFF,DM, 2, 1,1);
    launch_gemm(stream, ffbf,DFF,0,0, wc2T+(long)l*DM*DFF,DFF,0,0, tmp,nullptr,DM,0,0, enc_c2b+l*DM, xb_f,DM,0,0, M,DM,DFF, 0, 1,1);
    layernorm_kernel<<<(unsigned)M,256,0,stream>>>(tmp, enc_g2+l*DM, enc_be2+l*DM, xb_f, xb_b);
  }
  layernorm_kernel<<<(unsigned)M,256,0,stream>>>(xb_f, norm_g, norm_b, xb_f, xb_b);

  // -------- heads --------
  launch_gemm(stream, xb_b,DM,0,0,   wfc3T,DM,0,0,        tloss,nullptr,LL,0,0, fc3_b, nullptr,0,0,0, M,LL,DM, 0, 1,1);
  launch_gemm(stream, xb_b,DM,0,0,   wfcoT,2*DM,0,0,      dec,nullptr,LL,0,0, fco_b, nullptr,0,0,0, M,LL,DM, 0, 1,1);
  launch_gemm(stream, xfre_b,DM,0,0, wfcoT+DM,2*DM,0,0,   dec,nullptr,LL,0,0, nullptr, dec,LL,0,0, M,LL,DM, 0, 1,1);

  // -------- denorm + transpose to [B, pred, N] x3 --------
  finalize_kernel<<<cdiv_i(TOTOUT,256),256,0,stream>>>(dec, tloss, floss, revin_w, revin_b, meanv, stdv, out);
}
